// BidirectionalConvMambaBlock_67740224192689
// MI455X (gfx1250) — compile-verified, matrix operand order checked by reading, not run on hardware
//
#include <hip/hip_runtime.h>
#include <hip/hip_bf16.h>
#include <math.h>

typedef __bf16 bf16;
typedef __attribute__((ext_vector_type(16))) __bf16 v16bf;
typedef __attribute__((ext_vector_type(8)))  __bf16 v8bf;
typedef __attribute__((ext_vector_type(8)))  float  v8f;

#define DEV __device__ __forceinline__

DEV float sigmoidf_(float x){ return 1.f/(1.f+__expf(-x)); }
DEV float siluf_(float x){ return x * sigmoidf_(x); }
DEV bf16  tobf(float f){ return (bf16)f; }
DEV float tof(bf16 h){ return (float)h; }

DEV float wave_sum(float v){
  #pragma unroll
  for (int m = 16; m > 0; m >>= 1) v += __shfl_xor(v, m, 32);
  return v;
}

DEV float block_sum(float v, float* sred){
  __syncthreads();
  int lane = threadIdx.x & 31, w = threadIdx.x >> 5;
  v = wave_sum(v);
  if (lane == 0) sred[w] = v;
  __syncthreads();
  float r = 0.f;
  if ((int)threadIdx.x < (int)(blockDim.x >> 5)) r = sred[threadIdx.x];
  if (w == 0){ r = wave_sum(r); if (lane == 0) sred[0] = r; }
  __syncthreads();
  return sred[0];
}

// ---------------- cast fp32 -> bf16 ----------------
__global__ void cast_bf16_k(const float* __restrict__ in, bf16* __restrict__ out, int n){
  int i = blockIdx.x * blockDim.x + threadIdx.x;
  if (i < n) out[i] = tobf(in[i]);
}

// ---------------- RMSNorm ----------------
__global__ void rmsnorm_k(const float* __restrict__ x, const float* __restrict__ w,
                          float* __restrict__ xn, bf16* __restrict__ xnb){
  __shared__ float sred[32];
  int row = blockIdx.x;
  const float* xr = x + (size_t)row * 512;
  float ss = 0.f;
  for (int i = threadIdx.x; i < 512; i += blockDim.x){ float v = xr[i]; ss += v*v; }
  ss = block_sum(ss, sred);
  float sc = rsqrtf(ss * (1.f/512.f) + 1.1920929e-07f);
  for (int i = threadIdx.x; i < 512; i += blockDim.x){
    float v = xr[i] * sc * w[i];
    xn [(size_t)row*512 + i] = v;
    xnb[(size_t)row*512 + i] = tobf(v);
  }
}

// ------- depthwise conv3 + LayerNorm + SiLU + residual -> ssm_in (bf16) -------
__global__ void convln_k(const float* __restrict__ xn, const float* __restrict__ lcw,
                         const float* __restrict__ lcb, const float* __restrict__ lnw,
                         const float* __restrict__ lnb, bf16* __restrict__ ssm){
  __shared__ float sconv[512];
  __shared__ float sred[32];
  int row = blockIdx.x;
  int l = row & 2047;
  size_t base = (size_t)(row - l) * 512;
  for (int d = threadIdx.x; d < 512; d += blockDim.x){
    float acc = lcb[d];
    #pragma unroll
    for (int k = 0; k < 3; ++k){
      int t = l - 1 + k;
      if (t >= 0 && t < 2048) acc += xn[base + (size_t)t*512 + d] * lcw[d*3 + k];
    }
    sconv[d] = acc;
  }
  __syncthreads();
  float s = 0.f, s2 = 0.f;
  for (int d = threadIdx.x; d < 512; d += blockDim.x){ float v = sconv[d]; s += v; s2 += v*v; }
  s  = block_sum(s,  sred);
  s2 = block_sum(s2, sred);
  float mean = s * (1.f/512.f);
  float var  = s2 * (1.f/512.f) - mean*mean;
  float rstd = rsqrtf(var + 1e-5f);
  for (int d = threadIdx.x; d < 512; d += blockDim.x){
    float v = (sconv[d] - mean) * rstd * lnw[d] + lnb[d];
    v = siluf_(v) + xn[(size_t)row*512 + d];
    ssm[(size_t)row*512 + d] = tobf(v);
  }
}

DEV v16bf frag16(const bf16* p){
  v8bf lo = *(const v8bf*)p;
  v8bf hi = *(const v8bf*)(p + 16);
  return __builtin_shufflevector(lo, hi, 0,1,2,3,4,5,6,7,8,9,10,11,12,13,14,15);
}

// ---- WMMA GEMM: C[M,N] = act(A[M,K] * W[N,K]^T + bias); wave tile = 16M x (NT*16)N ----
// All fragment loads of a k-step are hoisted ahead of the WMMA chain so the
// scheduler can clause them and retire WMMAs against partial load counts.
// ACT: 0 = f32 out, 1 = bf16 out, 2 = softplus->f32, 3 = gelu->bf16, 4 = dual f32+bf16
template<int ACT, int NT>
__global__ void gemm_k(const bf16* __restrict__ A, int lda,
                       const bf16* __restrict__ W, int ldw,
                       const float* __restrict__ bias,
                       float* __restrict__ Cf, bf16* __restrict__ Cb, int ldc,
                       int K){
  int wave = threadIdx.x >> 5;
  int lane = threadIdx.x & 31;
  int mt  = blockIdx.x * 4 + wave;           // 16-row M tile
  int nt0 = blockIdx.y * NT;                 // first 16-col N tile
  int hl = lane & 15;
  int kb = (lane >> 4) << 3;                 // lanes 16-31 hold K offset +8
  const bf16* Ap  = A + (size_t)(mt*16 + hl)  * lda + kb;
  const bf16* Wp0 = W + (size_t)(nt0*16 + hl) * ldw + kb;
  v8f zero = {};
  v8f acc[NT];
  #pragma unroll
  for (int j = 0; j < NT; ++j) acc[j] = zero;
  for (int k0 = 0; k0 < K; k0 += 32){
    // ---- load phase: A frag + all NT B frags ----
    v16bf a = frag16(Ap + k0);               // K = k0+kb..+7 | k0+16+kb..+7
    v16bf b[NT];
    #pragma unroll
    for (int j = 0; j < NT; ++j)
      b[j] = frag16(Wp0 + (size_t)j*16*ldw + k0);
    __builtin_prefetch(Ap + k0 + 32, 0, 1);
    __builtin_prefetch(Wp0 + k0 + 32, 0, 1);
    // ---- compute phase: NT back-to-back WMMAs sharing the A frag ----
    #pragma unroll
    for (int j = 0; j < NT; ++j)
      acc[j] = __builtin_amdgcn_wmma_f32_16x16x32_bf16(false, a, false, b[j], (short)0, acc[j], false, false);
  }
  int orow0 = mt*16 + ((lane >> 4) << 3);    // C layout: lanes 16-31 hold M rows +8
  #pragma unroll
  for (int j = 0; j < NT; ++j){
    int ocol = (nt0 + j)*16 + hl;
    float bv = bias ? bias[ocol] : 0.f;
    #pragma unroll
    for (int r = 0; r < 8; ++r){
      float v = acc[j][r] + bv;
      if (ACT == 2) v = (v > 20.f) ? v : log1pf(__expf(v));
      if (ACT == 3) v = 0.5f * v * (1.f + erff(v * 0.70710678118f));
      size_t o = (size_t)(orow0 + r) * ldc + ocol;
      if (ACT == 0 || ACT == 2 || ACT == 4) Cf[o] = v;
      if (ACT == 1 || ACT == 3 || ACT == 4) Cb[o] = tobf(v);
    }
  }
}

// ------- causal (fwd) + anticausal (bwd) depthwise conv4 + SiLU from proj -------
__global__ void conv4_k(const bf16* __restrict__ proj, const float* __restrict__ cvw,
                        const float* __restrict__ cvb, bf16* __restrict__ xmf,
                        bf16* __restrict__ xmb){
  size_t idx = (size_t)blockIdx.x * blockDim.x + threadIdx.x;   // B*L*DI
  int d = (int)(idx & 1023);
  size_t row = idx >> 10;
  int l = (int)(row & 2047);
  size_t base = (row - (size_t)l) * 2048 + d;
  float wk[4] = { cvw[d*4], cvw[d*4+1], cvw[d*4+2], cvw[d*4+3] };
  float accf = cvb[d], accb = cvb[d];
  #pragma unroll
  for (int k = 0; k < 4; ++k){
    int tf = l - 3 + k;
    if (tf >= 0)   accf += wk[k] * tof(proj[base + (size_t)tf*2048]);
    int tb = l + 3 - k;
    if (tb < 2048) accb += wk[k] * tof(proj[base + (size_t)tb*2048]);
  }
  xmf[row*1024 + d] = tobf(siluf_(accf));
  xmb[row*1024 + d] = tobf(siluf_(accb));
}

// ------- selective scan: one wave32 per (b,d); lane = state index n -------
__global__ void scan_k(const float* __restrict__ dt, const bf16* __restrict__ xm,
                       const float* __restrict__ xd, const bf16* __restrict__ proj,
                       const float* __restrict__ A_log, const float* __restrict__ Dp,
                       bf16* __restrict__ yg, int rev){
  int gw   = (int)((blockIdx.x * blockDim.x + threadIdx.x) >> 5);
  int lane = threadIdx.x & 31;
  int d = gw & 1023;
  int b = gw >> 10;
  float A   = -__expf(A_log[d*32 + lane]);
  float dpd = Dp[d];
  float h = 0.f;
  for (int s = 0; s < 2048; ++s){
    int t = rev ? (2047 - s) : s;
    size_t row = (size_t)b * 2048 + t;
    float dtv = dt[row*1024 + d];
    float xv  = tof(xm[row*1024 + d]);
    float Bv  = xd[row*96 + 32 + lane];
    float Cv  = xd[row*96 + 64 + lane];
    h = __expf(dtv * A) * h + (dtv * xv) * Bv;
    float yc = wave_sum(h * Cv);
    if (lane == 0){
      float z = tof(proj[row*2048 + 1024 + d]);
      yg[row*1024 + d] = tobf((yc + dpd * xv) * (z * sigmoidf_(z)));
    }
  }
}

// ------- LayerNorm over concat [y_f | y_b] (1024) -> xs f32 -------
__global__ void ln_concat_k(const float* __restrict__ yf, const float* __restrict__ yb,
                            const float* __restrict__ w, const float* __restrict__ bsv,
                            float* __restrict__ xs){
  __shared__ float sc[1024];
  __shared__ float sred[32];
  int row = blockIdx.x;
  for (int c = threadIdx.x; c < 1024; c += blockDim.x)
    sc[c] = (c < 512) ? yf[(size_t)row*512 + c] : yb[(size_t)row*512 + (c - 512)];
  __syncthreads();
  float s = 0.f, s2 = 0.f;
  for (int c = threadIdx.x; c < 1024; c += blockDim.x){ float v = sc[c]; s += v; s2 += v*v; }
  s  = block_sum(s,  sred);
  s2 = block_sum(s2, sred);
  float mean = s * (1.f/1024.f);
  float var  = s2 * (1.f/1024.f) - mean*mean;
  float rstd = rsqrtf(var + 1e-5f);
  for (int c = threadIdx.x; c < 1024; c += blockDim.x)
    xs[(size_t)row*1024 + c] = (sc[c] - mean) * rstd * w[c] + bsv[c];
}

// ------- grouped conv3 (2 in-ch per group) + SiLU -> mixer -------
__global__ void mixer_k(const float* __restrict__ xs, const float* __restrict__ pcw,
                        const float* __restrict__ pcb, float* __restrict__ mixer){
  size_t idx = (size_t)blockIdx.x * blockDim.x + threadIdx.x;   // B*L*512
  int d = (int)(idx & 511);
  size_t row = idx >> 9;
  int l = (int)(row & 2047);
  size_t base = (row - (size_t)l) * 1024;
  float acc = pcb[d];
  #pragma unroll
  for (int c = 0; c < 2; ++c)
    #pragma unroll
    for (int k = 0; k < 3; ++k){
      int t = l - 1 + k;
      if (t >= 0 && t < 2048) acc += pcw[(d*2 + c)*3 + k] * xs[base + (size_t)t*1024 + 2*d + c];
    }
  mixer[row*512 + d] = siluf_(acc);
}

// ------- final residual: out = x + mixer + mlp -------
__global__ void final_k(const float* __restrict__ x, const float* __restrict__ mixer,
                        const float* __restrict__ mlp, float* __restrict__ out, int n){
  int i = blockIdx.x * blockDim.x + threadIdx.x;
  if (i < n) out[i] = x[i] + mixer[i] + mlp[i];
}

extern "C" void kernel_launch(void* const* d_in, const int* in_sizes, int n_in,
                              void* d_out, int out_size, void* d_ws, size_t ws_size,
                              hipStream_t stream){
  (void)in_sizes; (void)n_in; (void)out_size; (void)ws_size;
  const float* x     = (const float*)d_in[0];
  const float* norm_w= (const float*)d_in[1];
  const float* lc_w  = (const float*)d_in[2];
  const float* lc_b  = (const float*)d_in[3];
  const float* lnc_w = (const float*)d_in[4];
  const float* lnc_b = (const float*)d_in[5];
  const float* in_w  = (const float*)d_in[6];
  const float* cv_w  = (const float*)d_in[7];
  const float* cv_b  = (const float*)d_in[8];
  const float* xp_w  = (const float*)d_in[9];
  const float* dtp_w = (const float*)d_in[10];
  const float* dtp_b = (const float*)d_in[11];
  const float* A_log = (const float*)d_in[12];
  const float* Dp    = (const float*)d_in[13];
  const float* out_w = (const float*)d_in[14];
  const float* lnp_w = (const float*)d_in[15];
  const float* lnp_b = (const float*)d_in[16];
  const float* pc_w  = (const float*)d_in[17];
  const float* pc_b  = (const float*)d_in[18];
  const float* w1    = (const float*)d_in[19];
  const float* b1    = (const float*)d_in[20];
  const float* w2    = (const float*)d_in[21];
  const float* b2    = (const float*)d_in[22];

  constexpr int Rows = 4 * 2048;  // B*L
  char* wp = (char*)d_ws;
  auto alloc = [&](size_t bytes){ char* p = wp; wp += (bytes + 255) & ~(size_t)255; return p; };

  float* xn   = (float*)alloc((size_t)Rows*512*4);
  bf16*  xnb  = (bf16*) alloc((size_t)Rows*512*2);
  bf16*  ssm  = (bf16*) alloc((size_t)Rows*512*2);
  char*  projRaw =      alloc((size_t)Rows*2048*2);  // bf16 proj; later reused as xs f32 (same bytes)
  bf16*  proj = (bf16*)projRaw;
  bf16*  xmf  = (bf16*) alloc((size_t)Rows*1024*2);
  bf16*  xmb  = (bf16*) alloc((size_t)Rows*1024*2);
  float* xdf  = (float*)alloc((size_t)Rows*96*4);
  float* xdb  = (float*)alloc((size_t)Rows*96*4);
  bf16*  xdfb = (bf16*) alloc((size_t)Rows*96*2);
  bf16*  xdbb = (bf16*) alloc((size_t)Rows*96*2);
  char*  dtfRaw =       alloc((size_t)Rows*1024*4);  // f32 dt_f; later reused as h1 bf16 (same bytes)
  char*  dtbRaw =       alloc((size_t)Rows*1024*4);  // f32 dt_b; later reused as mlp_out f32
  float* dtf  = (float*)dtfRaw;
  float* dtb  = (float*)dtbRaw;
  bf16*  ygf  = (bf16*) alloc((size_t)Rows*1024*2);
  bf16*  ygb  = (bf16*) alloc((size_t)Rows*1024*2);
  char*  ypfRaw =       alloc((size_t)Rows*512*4);   // f32 yproj_f; later reused as mixer f32
  float* ypf  = (float*)ypfRaw;
  float* ypb  = (float*)alloc((size_t)Rows*512*4);
  bf16* in_wb  = (bf16*)alloc((size_t)2048*512*2);
  bf16* xp_wb  = (bf16*)alloc((size_t)96*1024*2);
  bf16* dtp_wb = (bf16*)alloc((size_t)1024*32*2);
  bf16* out_wb = (bf16*)alloc((size_t)512*1024*2);
  bf16* w1b    = (bf16*)alloc((size_t)2048*512*2);
  bf16* w2b    = (bf16*)alloc((size_t)512*2048*2);
  float* xsbuf = (float*)projRaw;
  bf16*  h1    = (bf16*)dtfRaw;
  float* mlpo  = (float*)dtbRaw;
  float* mixer = (float*)ypfRaw;

  auto cast = [&](const float* src, bf16* dst, int n){
    cast_bf16_k<<<(n + 255)/256, 256, 0, stream>>>(src, dst, n);
  };
  cast(in_w,  in_wb,  2048*512);
  cast(xp_w,  xp_wb,  96*1024);
  cast(dtp_w, dtp_wb, 1024*32);
  cast(out_w, out_wb, 512*1024);
  cast(w1,    w1b,    2048*512);
  cast(w2,    w2b,    512*2048);

  rmsnorm_k<<<Rows, 256, 0, stream>>>(x, norm_w, xn, xnb);
  convln_k <<<Rows, 256, 0, stream>>>(xn, lc_w, lc_b, lnc_w, lnc_b, ssm);

  dim3 gB(128);  // 4 waves -> 64 rows x (NT*16) cols per block
  // in-proj: [8192,512] x [2048,512]^T -> proj bf16 [8192,2048]
  gemm_k<1,4><<<dim3(Rows/64, 2048/64), gB, 0, stream>>>(ssm, 512, in_wb, 512, nullptr, nullptr, proj, 2048, 512);
  conv4_k<<<(Rows*1024)/256, 256, 0, stream>>>(proj, cv_w, cv_b, xmf, xmb);
  // x_dbl: [8192,1024] x [96,1024]^T -> f32 + bf16  (NT=3 -> 48-col blocks)
  gemm_k<4,3><<<dim3(Rows/64, 96/48), gB, 0, stream>>>(xmf, 1024, xp_wb, 1024, nullptr, xdf, xdfb, 96, 1024);
  gemm_k<4,3><<<dim3(Rows/64, 96/48), gB, 0, stream>>>(xmb, 1024, xp_wb, 1024, nullptr, xdb, xdbb, 96, 1024);
  // dt: [8192,32] x [1024,32]^T + dtp_b -> softplus (A strided in x_dbl, lda=96)
  gemm_k<2,4><<<dim3(Rows/64, 1024/64), gB, 0, stream>>>(xdfb, 96, dtp_wb, 32, dtp_b, dtf, nullptr, 1024, 32);
  gemm_k<2,4><<<dim3(Rows/64, 1024/64), gB, 0, stream>>>(xdbb, 96, dtp_wb, 32, dtp_b, dtb, nullptr, 1024, 32);
  // selective scans (fwd / bwd), fused output gating
  scan_k<<<(4*1024)/8, 256, 0, stream>>>(dtf, xmf, xdf, proj, A_log, Dp, ygf, 0);
  scan_k<<<(4*1024)/8, 256, 0, stream>>>(dtb, xmb, xdb, proj, A_log, Dp, ygb, 1);
  // out-proj: [8192,1024] x [512,1024]^T -> f32
  gemm_k<0,4><<<dim3(Rows/64, 512/64), gB, 0, stream>>>(ygf, 1024, out_wb, 1024, nullptr, ypf, nullptr, 512, 1024);
  gemm_k<0,4><<<dim3(Rows/64, 512/64), gB, 0, stream>>>(ygb, 1024, out_wb, 1024, nullptr, ypb, nullptr, 512, 1024);
  ln_concat_k<<<Rows, 256, 0, stream>>>(ypf, ypb, lnp_w, lnp_b, xsbuf);
  mixer_k<<<(Rows*512)/256, 256, 0, stream>>>(xsbuf, pc_w, pc_b, mixer);
  // MLP: gelu(xn @ w1^T + b1) @ w2^T + b2
  gemm_k<3,4><<<dim3(Rows/64, 2048/64), gB, 0, stream>>>(xnb, 512, w1b, 512, b1, nullptr, h1, 2048, 512);
  gemm_k<0,4><<<dim3(Rows/64, 512/64), gB, 0, stream>>>(h1, 2048, w2b, 2048, b2, mlpo, nullptr, 512, 2048);
  final_k<<<(Rows*512)/256, 256, 0, stream>>>(x, mixer, mlpo, (float*)d_out, Rows*512);
}
